// MultiHeadAttention_38311108281137
// MI455X (gfx1250) — matrix operand order repaired and checked by reading, not compiled, not run on hardware
//
#include <hip/hip_runtime.h>
#include <stdint.h>

// MHA for MI455X (gfx1250, wave32, WMMA f32_16x16x32_f16).
// Pipeline: transpose/convert weights -> 3 proj GEMMs -> flash attention -> out GEMM.
// Pure-copy tiles staged with GLOBAL_LOAD_ASYNC_TO_LDS_B128 (ASYNCcnt), double-buffered
// so the async DMA of tile k+1 overlaps the WMMA work on tile k.

typedef _Float16 half_t;
typedef __attribute__((ext_vector_type(16))) _Float16 v16h;
typedef __attribute__((ext_vector_type(8)))  _Float16 v8h;
typedef __attribute__((ext_vector_type(8)))  float    v8f;

#define D_MODEL   1024
#define NUM_HEADS 16
#define DEPTH     64
#define BATCH     4
#define SEQ       2048
#define NEGBIG    (-1e30f)

// ---------- async global->LDS copy (16 bytes per active lane) ----------

__device__ __forceinline__ void async_g2l_b128(half_t* lds, const half_t* g) {
  // LDS byte offset = low 32 bits of the generic (flat) address of a __shared__ object.
  uint32_t l = (uint32_t)(uintptr_t)lds;
  asm volatile("global_load_async_to_lds_b128 %0, %1, off"
               :: "v"(l), "v"(g) : "memory");
}

__device__ __forceinline__ void async_wait0() {
#if __has_builtin(__builtin_amdgcn_s_wait_asynccnt)
  __builtin_amdgcn_s_wait_asynccnt(0);
#else
  asm volatile("s_wait_asynccnt 0" ::: "memory");
#endif
}

// ---------- fragment helpers (this code's K order hh*16+e, used for both A and B) ----------

// 16 consecutive f16 at p -> B-fragment slice (lane holds col=lane&15, K=(lane>=16)*16+e)
__device__ __forceinline__ v16h ld16(const half_t* p) {
  v8h lo = *(const v8h*)p;
  v8h hi = *(const v8h*)(p + 8);
  v16h o;
#pragma unroll
  for (int i = 0; i < 8; ++i) { o[i] = lo[i]; o[i + 8] = hi[i]; }
  return o;
}

// A-fragment (16x32 f16): lane holds row r=lane&15; elems 0..15 = K hh*16+0..15.
// 'row' = pointer to this lane's row (32-chunk base).
__device__ __forceinline__ v16h ld_afrag(const half_t* row, int hh) {
  v8h lo = *(const v8h*)(row + hh * 16);
  v8h hi = *(const v8h*)(row + hh * 16 + 8);
  v16h o;
#pragma unroll
  for (int i = 0; i < 8; ++i) { o[i] = lo[i]; o[i + 8] = hi[i]; }
  return o;
}

__device__ __forceinline__ v8f wmma_f16(v16h a, v16h b, v8f c) {
  return __builtin_amdgcn_wmma_f32_16x16x32_f16(false, a, false, b, (short)0, c,
                                                false, false);
}

// reductions across the 16 lanes of a half-wave (rows live per-half in C layout)
__device__ __forceinline__ float redmax16(float v) {
#pragma unroll
  for (int m = 1; m < 16; m <<= 1) v = fmaxf(v, __shfl_xor(v, m, 32));
  return v;
}
__device__ __forceinline__ float redsum16(float v) {
#pragma unroll
  for (int m = 1; m < 16; m <<= 1) v += __shfl_xor(v, m, 32);
  return v;
}

// ---------- kernel 1: weight transpose + f16 convert: Wt[n][k] = (f16)W[k][n] ----------

__global__ __launch_bounds__(256) void k_transpose_w(const float* __restrict__ W,
                                                     half_t* __restrict__ Wt) {
  int base = (blockIdx.x * 256 + threadIdx.x) * 4;   // 16B-aligned coalesced read
  float4 w4 = *(const float4*)(W + base);
  int k = base >> 10, n = base & 1023;
  Wt[(size_t)(n + 0) * D_MODEL + k] = (half_t)w4.x;
  Wt[(size_t)(n + 1) * D_MODEL + k] = (half_t)w4.y;
  Wt[(size_t)(n + 2) * D_MODEL + k] = (half_t)w4.z;
  Wt[(size_t)(n + 3) * D_MODEL + k] = (half_t)w4.w;
}

// ---------- kernel 2: projection GEMM  Y[b,h,s,d] = (A[m,:] @ W[:,n] + bias[n]) * scale ----------
// A fp32 [8192 x 1024]; Wt f16 [n][k]; block tile 128(M) x 64(N) = one head; 8 waves x 16 rows.

__global__ __launch_bounds__(256)
void k_proj(const float* __restrict__ A, const half_t* __restrict__ Wt,
            const float* __restrict__ bias, half_t* __restrict__ Y, float scaleOut) {
  __shared__ __align__(16) half_t Alds[2][128 * 32];
  __shared__ __align__(16) half_t Blds[2][64 * 32];
  const int tid = threadIdx.x;
  const int lane = tid & 31, wave = tid >> 5;
  const int c = lane & 15, hh = lane >> 4;
  const int m0 = blockIdx.y * 128;
  const int h  = blockIdx.x;              // N tile == one head (64 cols)
  const int n0 = h * 64;
  const int bn = tid >> 2, bkk = (tid & 3) * 8;   // B-tile async coords
  const half_t* Wrow = Wt + (size_t)(n0 + bn) * D_MODEL + bkk;
  v8f acc[4] = {};

  // prologue: async-stage B tile 0 (2048 halfs = 256 x 16B DMA)
  async_g2l_b128(&Blds[0][tid * 8], Wrow);

  for (int kt = 0; kt < D_MODEL / 32; ++kt) {
    const int k0 = kt * 32;
    const int cur = kt & 1, nxt = cur ^ 1;
    // register path: A tile fp32 -> f16 into Alds[cur]
#pragma unroll
    for (int it = 0; it < 2; ++it) {
      int idx8 = (tid + it * 256) * 8;
      int r = idx8 >> 5, kk = idx8 & 31;
      const float4* src = (const float4*)(A + (size_t)(m0 + r) * D_MODEL + k0 + kk);
      float4 f0 = src[0], f1 = src[1];
      v8h hv;
      hv[0] = (half_t)f0.x; hv[1] = (half_t)f0.y; hv[2] = (half_t)f0.z; hv[3] = (half_t)f0.w;
      hv[4] = (half_t)f1.x; hv[5] = (half_t)f1.y; hv[6] = (half_t)f1.z; hv[7] = (half_t)f1.w;
      *(v8h*)&Alds[cur][idx8] = hv;
    }
    async_wait0();          // B tile kt has landed
    __syncthreads();        // publish tile kt; Blds[nxt] free for overwrite
    if (kt + 1 < D_MODEL / 32)  // overlap next B-tile DMA with this tile's WMMAs
      async_g2l_b128(&Blds[nxt][tid * 8], Wrow + k0 + 32);
    v16h a = ld_afrag(&Alds[cur][(wave * 16 + c) * 32], hh);
#pragma unroll
    for (int nb = 0; nb < 4; ++nb) {
      v16h b = ld16(&Blds[cur][(nb * 16 + c) * 32 + hh * 16]);
      acc[nb] = wmma_f16(a, b, acc[nb]);
    }
  }
#pragma unroll
  for (int nb = 0; nb < 4; ++nb) {
    int d = nb * 16 + c;
    float bv = bias[n0 + d];
#pragma unroll
    for (int i = 0; i < 8; ++i) {
      int m = m0 + wave * 16 + i + hh * 8;
      int b = m >> 11, s = m & 2047;
      float val = (acc[nb][i] + bv) * scaleOut;
      Y[(((size_t)(b * NUM_HEADS + h) * SEQ + s) * DEPTH) + d] = (half_t)val;
    }
  }
}

// ---------- kernel 3: flash attention ----------
// grid (S/128, B*H), 8 waves x 16 query rows. 32-key blocks, K tile via async DMA
// (double-buffered), V tile transposed through registers, online softmax in fp32.

__global__ __launch_bounds__(256)
void k_flash(const half_t* __restrict__ Qh, const half_t* __restrict__ Kh,
             const half_t* __restrict__ Vh, const int* __restrict__ mask,
             half_t* __restrict__ Oh) {
  __shared__ __align__(16) half_t Klds[2][32 * 64];   // [key][depth]
  __shared__ __align__(16) half_t Vtlds[2][64 * 32];  // [depth][key] (transposed)
  __shared__ __align__(16) half_t Pbuf[8][16 * 32];   // per-wave P transpose buffer
  __shared__ float maskneg[2][32];
  const int tid = threadIdx.x;
  const int lane = tid & 31, wave = tid >> 5;
  const int c = lane & 15, hh = lane >> 4;
  const int bh = blockIdx.y;
  const int b = bh >> 4, h = bh & 15;
  const int s0 = blockIdx.x * 128;
  const size_t headBase = (size_t)bh * SEQ * DEPTH;
  const int NKB = SEQ / 32;

  // Q fragments (rows s0+wave*16+r, depth chunks 0..31 / 32..63); 1/8 scale folded in.
  const half_t* qrow = Qh + headBase + (size_t)(s0 + wave * 16 + c) * DEPTH;
  v16h aq0 = ld_afrag(qrow, hh);
  v16h aq1 = ld_afrag(qrow + 32, hh);

  v8f o[4] = {};
  float mrow[8], lrow[8];
#pragma unroll
  for (int i = 0; i < 8; ++i) { mrow[i] = -__builtin_inff(); lrow[i] = 0.0f; }

  // prologue: async-stage K tile 0 (4KB = 256 x 16B DMA)
  async_g2l_b128(&Klds[0][tid * 8], Kh + headBase + tid * 8);

  for (int kb = 0; kb < NKB; ++kb) {
    const int cur = kb & 1, nxt = cur ^ 1;
    const half_t* Vg = Vh + headBase + (size_t)kb * 32 * DEPTH;
    // register path: V tile transpose into Vtlds[cur]
    {
      int j = tid >> 3, d0 = (tid & 7) * 8;
      v8h vv = *(const v8h*)(Vg + j * 64 + d0);
#pragma unroll
      for (int e = 0; e < 8; ++e) Vtlds[cur][(d0 + e) * 32 + j] = vv[e];
    }
    if (tid < 32) maskneg[cur][tid] = (float)mask[b * SEQ + kb * 32 + tid] * NEGBIG;
    if (kb + 1 < NKB)                      // global_prefetch_b8 for next V tile
      __builtin_prefetch(Vg + 32 * DEPTH + tid * 8, 0, 0);
    async_wait0();          // K tile kb has landed
    __syncthreads();        // publish tile kb; Klds[nxt] free for overwrite
    if (kb + 1 < NKB)       // overlap next K-tile DMA with this tile's compute
      async_g2l_b128(&Klds[nxt][tid * 8],
                     Kh + headBase + (size_t)(kb + 1) * 32 * DEPTH + tid * 8);

    // scores: 16 queries x 32 keys (two 16-col C fragments), contraction depth=64
    v8f sc0 = {}, sc1 = {};
    {
      const half_t* krow0 = &Klds[cur][c * 64];
      const half_t* krow1 = &Klds[cur][(16 + c) * 64];
      sc0 = wmma_f16(aq0, ld16(krow0 + hh * 16), sc0);
      sc0 = wmma_f16(aq1, ld16(krow0 + 32 + hh * 16), sc0);
      sc1 = wmma_f16(aq0, ld16(krow1 + hh * 16), sc1);
      sc1 = wmma_f16(aq1, ld16(krow1 + 32 + hh * 16), sc1);
    }
    float mn0 = maskneg[cur][c], mn1 = maskneg[cur][16 + c];
#pragma unroll
    for (int i = 0; i < 8; ++i) {
      float x0 = sc0[i] + mn0;
      float x1 = sc1[i] + mn1;
      float vmax = redmax16(fmaxf(x0, x1));
      float mnew = fmaxf(mrow[i], vmax);
      float p0 = __expf(x0 - mnew);
      float p1 = __expf(x1 - mnew);
      float corr = __expf(mrow[i] - mnew);
      float psum = redsum16(p0 + p1);
      lrow[i] = lrow[i] * corr + psum;
      mrow[i] = mnew;
#pragma unroll
      for (int db = 0; db < 4; ++db) o[db][i] *= corr;
      int prow = i + hh * 8;             // C-layout row -> LDS (transpose via LDS)
      Pbuf[wave][prow * 32 + c]      = (half_t)p0;
      Pbuf[wave][prow * 32 + 16 + c] = (half_t)p1;
    }
    // O += P(16x32) @ V(32x64); same-wave DS ops are in-order, no barrier needed
    v16h pfrag = ld_afrag(&Pbuf[wave][c * 32], hh);
#pragma unroll
    for (int db = 0; db < 4; ++db) {
      const half_t* vrow = &Vtlds[cur][(db * 16 + c) * 32];
      o[db] = wmma_f16(pfrag, ld16(vrow + hh * 16), o[db]);
    }
  }

  float inv[8];
#pragma unroll
  for (int i = 0; i < 8; ++i) inv[i] = 1.0f / lrow[i];
#pragma unroll
  for (int db = 0; db < 4; ++db) {
    int d = db * 16 + c;
#pragma unroll
    for (int i = 0; i < 8; ++i) {
      int s = s0 + wave * 16 + i + hh * 8;
      Oh[((size_t)(b * SEQ + s)) * D_MODEL + h * DEPTH + d] =
          (half_t)(o[db][i] * inv[i]);
    }
  }
}

// ---------- kernel 4: output GEMM  out[m,n] = attn[m,:] @ wo[:,n] + bo[n] (fp32 out) ----------
// All staging is pure f16 copy -> fully async, double-buffered.

__global__ __launch_bounds__(256)
void k_out(const half_t* __restrict__ A, const half_t* __restrict__ Wt,
           const float* __restrict__ bias, float* __restrict__ Out) {
  __shared__ __align__(16) half_t Alds[2][128 * 32];
  __shared__ __align__(16) half_t Blds[2][64 * 32];
  const int tid = threadIdx.x;
  const int lane = tid & 31, wave = tid >> 5;
  const int c = lane & 15, hh = lane >> 4;
  const int m0 = blockIdx.y * 128;
  const int n0 = blockIdx.x * 64;
  const int bn = tid >> 2, bkk = (tid & 3) * 8;
  const half_t* Wrow = Wt + (size_t)(n0 + bn) * D_MODEL + bkk;
  v8f acc[4] = {};

  // prologue: async-stage tile 0 (A: 2 x 16B per thread, B: 1 x 16B per thread)
#pragma unroll
  for (int it = 0; it < 2; ++it) {
    int idx8 = (tid + it * 256) * 8;
    int r = idx8 >> 5, kk = idx8 & 31;
    async_g2l_b128(&Alds[0][idx8], A + (size_t)(m0 + r) * D_MODEL + kk);
  }
  async_g2l_b128(&Blds[0][tid * 8], Wrow);

  for (int kt = 0; kt < D_MODEL / 32; ++kt) {
    const int k0 = kt * 32;
    const int cur = kt & 1, nxt = cur ^ 1;
    async_wait0();          // tile kt has landed
    __syncthreads();        // publish tile kt; buffers[nxt] free for overwrite
    if (kt + 1 < D_MODEL / 32) {  // overlap next tile's DMA with this tile's WMMAs
#pragma unroll
      for (int it = 0; it < 2; ++it) {
        int idx8 = (tid + it * 256) * 8;
        int r = idx8 >> 5, kk = idx8 & 31;
        async_g2l_b128(&Alds[nxt][idx8],
                       A + (size_t)(m0 + r) * D_MODEL + k0 + 32 + kk);
      }
      async_g2l_b128(&Blds[nxt][tid * 8], Wrow + k0 + 32);
    }
    v16h a = ld_afrag(&Alds[cur][(wave * 16 + c) * 32], hh);
#pragma unroll
    for (int nb = 0; nb < 4; ++nb) {
      v16h b = ld16(&Blds[cur][(nb * 16 + c) * 32 + hh * 16]);
      acc[nb] = wmma_f16(a, b, acc[nb]);
    }
  }
#pragma unroll
  for (int nb = 0; nb < 4; ++nb) {
    int n = n0 + nb * 16 + c;
    float bv = bias[n];
#pragma unroll
    for (int i = 0; i < 8; ++i) {
      int m = m0 + wave * 16 + i + hh * 8;
      Out[(size_t)m * D_MODEL + n] = acc[nb][i] + bv;
    }
  }
}

// ---------- launch ----------

extern "C" void kernel_launch(void* const* d_in, const int* in_sizes, int n_in,
                              void* d_out, int out_size, void* d_ws, size_t ws_size,
                              hipStream_t stream) {
  (void)in_sizes; (void)n_in; (void)out_size; (void)ws_size;
  const float* v    = (const float*)d_in[0];
  const float* k    = (const float*)d_in[1];
  const float* q    = (const float*)d_in[2];
  const int*   mask = (const int*)d_in[3];
  const float* wq = (const float*)d_in[4];
  const float* bq = (const float*)d_in[5];
  const float* wk = (const float*)d_in[6];
  const float* bk = (const float*)d_in[7];
  const float* wv = (const float*)d_in[8];
  const float* bv = (const float*)d_in[9];
  const float* wo = (const float*)d_in[10];
  const float* bo = (const float*)d_in[11];
  float* out = (float*)d_out;

  // workspace layout (f16): qh | kh | vh | 4 transposed weights | attn   (~72 MB)
  const size_t NH = (size_t)BATCH * NUM_HEADS * SEQ * DEPTH;   // 8,388,608
  const size_t WSZ = (size_t)D_MODEL * D_MODEL;                // 1,048,576
  half_t* qh   = (half_t*)d_ws;
  half_t* kh   = qh + NH;
  half_t* vh   = kh + NH;
  half_t* wt   = vh + NH;
  half_t* attn = wt + 4 * WSZ;

  k_transpose_w<<<1024, 256, 0, stream>>>(wq, wt + 0 * WSZ);
  k_transpose_w<<<1024, 256, 0, stream>>>(wk, wt + 1 * WSZ);
  k_transpose_w<<<1024, 256, 0, stream>>>(wv, wt + 2 * WSZ);
  k_transpose_w<<<1024, 256, 0, stream>>>(wo, wt + 3 * WSZ);

  dim3 gproj(NUM_HEADS, (BATCH * SEQ) / 128);   // (16, 64)
  const float qscale = 0.125f;                  // 1/sqrt(DEPTH), folded into Q
  k_proj<<<gproj, 256, 0, stream>>>(q, wt + 0 * WSZ, bq, qh, qscale);
  k_proj<<<gproj, 256, 0, stream>>>(k, wt + 1 * WSZ, bk, kh, 1.0f);
  k_proj<<<gproj, 256, 0, stream>>>(v, wt + 2 * WSZ, bv, vh, 1.0f);

  dim3 gfa(SEQ / 128, BATCH * NUM_HEADS);       // (16, 64)
  k_flash<<<gfa, 256, 0, stream>>>(qh, kh, vh, mask, attn);

  dim3 gout(D_MODEL / 64, (BATCH * SEQ) / 128); // (16, 64)
  k_out<<<gout, 256, 0, stream>>>(attn, wt + 3 * WSZ, bo, out);
}
